// GCN1_90881507983767
// MI455X (gfx1250) — matrix operand order repaired and checked by reading, not compiled, not run on hardware
//
#include <hip/hip_runtime.h>
#include <cstdint>
#include <cstddef>

#define DF 128

typedef __attribute__((ext_vector_type(16))) __bf16 v16bf;
typedef __attribute__((ext_vector_type(2)))  __bf16 v2bf;
typedef __attribute__((ext_vector_type(8)))  float  v8f;

// Native packed f32->bf16 convert (v_cvt_pk_bf16_f32) when the builtin exists;
// otherwise plain casts (clang lowers fptrunc->bf16 natively on gfx1250).
__device__ __forceinline__ v2bf pk2(float lo, float hi) {
#if __has_builtin(__builtin_amdgcn_cvt_pk_bf16_f32)
  return __builtin_amdgcn_cvt_pk_bf16_f32(lo, hi);
#else
  v2bf r; r[0] = (__bf16)lo; r[1] = (__bf16)hi; return r;
#endif
}

__device__ __forceinline__ __bf16 f2bf(float f) { return (__bf16)f; }

__device__ __forceinline__ void atomic_add_f32(float* p, float v) {
  (void)__hip_atomic_fetch_add(p, v, __ATOMIC_RELAXED, __HIP_MEMORY_SCOPE_AGENT);
}

// ---- setup kernels -------------------------------------------------------

// Pack W (f32, [K=in,128][N=out,128]) into the WMMA B-fragment layout so each
// lane's 16-bf16 fragment is one contiguous 32B load.
// idx = ((t*4 + s)*32 + lane)*16 + e ; K = s*32 + (lane>=16 ? 8:0) + (e&7) + (e>=8 ? 16:0) ; N = t*16 + (lane&15)
__global__ void pack_w(const float* __restrict__ W, __bf16* __restrict__ Wp) {
  int idx = blockIdx.x * blockDim.x + threadIdx.x;
  if (idx >= DF * DF) return;
  int e = idx & 15;
  int l = (idx >> 4) & 31;
  int s = (idx >> 9) & 3;
  int t = idx >> 11;
  int k  = s * 32 + ((l & 16) ? 8 : 0) + (e & 7) + ((e & 8) ? 16 : 0);
  int nn = t * 16 + (l & 15);
  Wp[idx] = f2bf(W[k * DF + nn]);
}

__global__ void deg_init(float* __restrict__ deg, int n) {
  int i = blockIdx.x * blockDim.x + threadIdx.x;
  if (i < n) deg[i] = 2.0f;               // improved=True self-loop weight
}

__global__ void deg_edge(const int* __restrict__ dst, float* __restrict__ deg, int E) {
  int i = blockIdx.x * blockDim.x + threadIdx.x;
  if (i < E) atomic_add_f32(&deg[dst[i]], 1.0f);
}

__global__ void dinv_k(float* __restrict__ dinv, int n) {
  int i = blockIdx.x * blockDim.x + threadIdx.x;
  if (i < n) dinv[i] = rsqrtf(dinv[i]);   // deg >= 2 always
}

__global__ void norm_k(const int* __restrict__ src, const int* __restrict__ dst,
                       const float* __restrict__ dinv, float* __restrict__ norm, int E) {
  int i = blockIdx.x * blockDim.x + threadIdx.x;
  if (i < E) norm[i] = dinv[src[i]] * dinv[dst[i]];
}

// ---- GEMM: Out[n x 128] = H[n x 128] @ W (bf16 WMMA, f32 accumulate) -----
// Block = 256 threads = 8 waves; each wave computes a 16-row x 128-col strip.
__global__ __launch_bounds__(256) void gemm_bf16(const float* __restrict__ H,
                                                 const __bf16* __restrict__ Wp,
                                                 float* __restrict__ Out, int n) {
  const int wid  = threadIdx.x >> 5;
  const int lane = threadIdx.x & 31;
  const int half = lane >> 4;             // 0 or 1
  const int l15  = lane & 15;
  const int rowBase = blockIdx.x * 128 + wid * 16;

  int arow = rowBase + l15;               // A-matrix: M = lane&15 for both halves
  if (arow >= n) arow = n - 1;            // clamp (stores are guarded)

  v8f c[8];
#pragma unroll
  for (int t = 0; t < 8; ++t) c[t] = (v8f){0.f, 0.f, 0.f, 0.f, 0.f, 0.f, 0.f, 0.f};

  const v16bf* bp = (const v16bf*)Wp;

#pragma unroll
  for (int s = 0; s < 4; ++s) {
    const int k0 = s * 32;
    // A fragment: elements 0..7 = K k0+off..+7, elements 8..15 = K k0+off+16..+23
    const float4* q = (const float4*)(H + (size_t)arow * DF + k0 + half * 8);
    float4 x0 = q[0];
    float4 x1 = q[1];
    float4 x2 = q[4];
    float4 x3 = q[5];

    union { v16bf v; v2bf p[8]; } a;
    a.p[0] = pk2(x0.x, x0.y);
    a.p[1] = pk2(x0.z, x0.w);
    a.p[2] = pk2(x1.x, x1.y);
    a.p[3] = pk2(x1.z, x1.w);
    a.p[4] = pk2(x2.x, x2.y);
    a.p[5] = pk2(x2.z, x2.w);
    a.p[6] = pk2(x3.x, x3.y);
    a.p[7] = pk2(x3.z, x3.w);

#pragma unroll
    for (int t = 0; t < 8; ++t) {
      v16bf b = bp[(t * 4 + s) * 32 + lane];
      c[t] = __builtin_amdgcn_wmma_f32_16x16x32_bf16(false, a.v, false, b,
                                                     (short)0, c[t], false, false);
    }
  }

  // D layout: VGPR r -> M = (lane>=16 ? 8:0)+r, N = lane&15
  const int orowBase = rowBase + half * 8;
#pragma unroll
  for (int t = 0; t < 8; ++t) {
    const int col = t * 16 + l15;
#pragma unroll
    for (int r = 0; r < 8; ++r) {
      int orow = orowBase + r;
      if (orow < n) Out[(size_t)orow * DF + col] = c[t][r];
    }
  }
}

// ---- Aggregation ---------------------------------------------------------

// out = (2/deg) * tmp (self-loop term) + bias  -- also zero-initializes out
__global__ void agg_init(const float* __restrict__ tmp, const float* __restrict__ dinv,
                         const float* __restrict__ bias, float* __restrict__ out, int n) {
  long idx = (long)blockIdx.x * blockDim.x + threadIdx.x;
  if (idx >= (long)n * DF) return;
  int node = (int)(idx >> 7);
  int dcol = (int)(idx & 127);
  float di = dinv[node];
  out[idx] = 2.0f * di * di * tmp[idx] + bias[dcol];
}

// One wave per edge: out[dst] += norm[e] * tmp[src]; 4 f32 atomics per lane.
// tmp (51.2 MB) and out are L2-resident, so both the gather and the atomic
// scatter are served by the 192 MB L2, not HBM.
__global__ __launch_bounds__(256) void agg_edge(const float* __restrict__ tmp,
                                                const int* __restrict__ src,
                                                const int* __restrict__ dst,
                                                const float* __restrict__ norm,
                                                float* __restrict__ out, int E) {
  int e = blockIdx.x * 8 + (threadIdx.x >> 5);
  if (e >= E) return;
  int lane = threadIdx.x & 31;
  int s = src[e];
  int d = dst[e];
  float nv = norm[e];
  float4 v = ((const float4*)(tmp + (size_t)s * DF))[lane];
  float* o = out + (size_t)d * DF + lane * 4;
  atomic_add_f32(o + 0, nv * v.x);
  atomic_add_f32(o + 1, nv * v.y);
  atomic_add_f32(o + 2, nv * v.z);
  atomic_add_f32(o + 3, nv * v.w);
}

// ---- driver --------------------------------------------------------------

extern "C" void kernel_launch(void* const* d_in, const int* in_sizes, int n_in,
                              void* d_out, int out_size, void* d_ws, size_t ws_size,
                              hipStream_t stream) {
  const float* x  = (const float*)d_in[0];
  const int*   ei = (const int*)d_in[1];
  const float* W1 = (const float*)d_in[2];
  const float* b1 = (const float*)d_in[3];
  const float* W2 = (const float*)d_in[4];
  const float* b2 = (const float*)d_in[5];
  const int n = in_sizes[0] / DF;
  const int E = in_sizes[1] / 2;
  const int* srcI = ei;        // edge_index[0]
  const int* dstI = ei + E;    // edge_index[1]
  float* out = (float*)d_out;

  char* w = (char*)d_ws;
  float* bufA = (float*)w; w += (size_t)n * DF * sizeof(float);
  float* bufB = (float*)w; w += (size_t)n * DF * sizeof(float);
  float* dinv = (float*)w; w += (((size_t)n * sizeof(float)) + 31) & ~(size_t)31;
  float* norm = (float*)w; w += (((size_t)E * sizeof(float)) + 31) & ~(size_t)31;
  __bf16* Wp1 = (__bf16*)w; w += (size_t)DF * DF * sizeof(__bf16);
  __bf16* Wp2 = (__bf16*)w;

  // weight prepack + symmetric GCN normalization
  pack_w<<<(DF * DF + 255) / 256, 256, 0, stream>>>(W1, Wp1);
  pack_w<<<(DF * DF + 255) / 256, 256, 0, stream>>>(W2, Wp2);
  deg_init<<<(n + 255) / 256, 256, 0, stream>>>(dinv, n);
  deg_edge<<<(E + 255) / 256, 256, 0, stream>>>(dstI, dinv, E);
  dinv_k<<<(n + 255) / 256, 256, 0, stream>>>(dinv, n);
  norm_k<<<(E + 255) / 256, 256, 0, stream>>>(srcI, dstI, dinv, norm, E);

  const int gblocks = (n + 127) / 128;
  const long tot = (long)n * DF;
  const int iblocks = (int)((tot + 255) / 256);
  const int eblocks = (E + 7) / 8;

  // conv1: x -> bufA (via bufB)
  gemm_bf16<<<gblocks, 256, 0, stream>>>(x, Wp1, bufB, n);
  agg_init<<<iblocks, 256, 0, stream>>>(bufB, dinv, b1, bufA, n);
  agg_edge<<<eblocks, 256, 0, stream>>>(bufB, srcI, dstI, norm, bufA, E);

  // conv2..conv4: bufA -> bufA (via bufB)
  for (int i = 0; i < 3; ++i) {
    gemm_bf16<<<gblocks, 256, 0, stream>>>(bufA, Wp2, bufB, n);
    agg_init<<<iblocks, 256, 0, stream>>>(bufB, dinv, b2, bufA, n);
    agg_edge<<<eblocks, 256, 0, stream>>>(bufB, srcI, dstI, norm, bufA, E);
  }

  // conv5: bufA -> d_out (via bufB)
  gemm_bf16<<<gblocks, 256, 0, stream>>>(bufA, Wp2, bufB, n);
  agg_init<<<iblocks, 256, 0, stream>>>(bufB, dinv, b2, out, n);
  agg_edge<<<eblocks, 256, 0, stream>>>(bufB, srcI, dstI, norm, out, E);
}
